// WeightedPoolingSpanExtractor_24300924961244
// MI455X (gfx1250) — matrix operand order repaired and checked by reading, not compiled, not run on hardware
//
#include <hip/hip_runtime.h>
#include <hip/hip_bf16.h>

// Problem constants (from reference)
#define B_DIM   8
#define T_DIM   512
#define E_DIM   1024
#define NSPANS  512
#define MAXW    32
#define RED     512            // REDUCED_DIM (embedding part)
#define WCOLS   513            // REDUCED_DIM + 1 (W_red columns)
#define WEMB_N  64
#define WEMB_D  128
#define OUT_D   (RED + WEMB_D) // 640
#define NEG_INF_F (-1000000000.0f)

#define KTILES  (E_DIM / 32)   // 32
#define NTILES  (RED / 16)     // 32
#define WFRAG_ELEMS ((size_t)KTILES * NTILES * 32 * 16)  // 524288

typedef __attribute__((ext_vector_type(16))) __bf16 v16bf;
typedef __attribute__((ext_vector_type(8)))  float  v8f;

// ---------------------------------------------------------------------------
// Kernel 0: one-shot conversion of W (E x 513 f32) into pre-swizzled bf16
// hi/lo WMMA B-fragments.  Layout: [kt][nt][lane][slot s], slot s of lane
// holds B[K = kt*32 + 8*(lane>>4) + (s&7) + 16*(s>>3), n = nt*16 + (lane&15)].
// ---------------------------------------------------------------------------
__global__ __launch_bounds__(256)
void wpse_prep_w(const float* __restrict__ W,
                 __bf16* __restrict__ whi,
                 __bf16* __restrict__ wlo)
{
    const int idx  = blockIdx.x * 256 + threadIdx.x;   // 0 .. 524287
    const int s    = idx & 15;
    const int lane = (idx >> 4) & 31;
    const int nt   = (idx >> 9) & 31;
    const int kt   = idx >> 14;
    const int K    = kt * 32 + 8 * (lane >> 4) + (s & 7) + 16 * (s >> 3);
    const int n    = nt * 16 + (lane & 15);
    const float x  = W[(size_t)K * WCOLS + n];
    const __bf16 h = (__bf16)x;
    whi[idx] = h;
    wlo[idx] = (__bf16)(x - (float)h);
}

// ---------------------------------------------------------------------------
// Kernel 1: reduced_emb[r, n] = sum_e seq[r, e] * W[e, n] + bias[n]
//   bf16x3 split-precision WMMA, f32 accumulate.
//   Per-wave output tile: 32(M) x 64(N) = 8 accumulators, so each B fragment
//   feeds 2 M-tiles and each A-split feeds 4 N-tiles.
//   Grid: 128 blocks of 8 waves.  ngroup = blockIdx>>4 shared by all waves in
//   the block (B fragments hit L0/L1); 32-row M-range differs per wave.
// ---------------------------------------------------------------------------
__global__ __launch_bounds__(256)
void wpse_gemm_wmma(const float*  __restrict__ seq,
                    const __bf16* __restrict__ whi,
                    const __bf16* __restrict__ wlo,
                    const float*  __restrict__ bias,
                    float*        __restrict__ red)
{
    const int lane  = threadIdx.x & 31;
    const int wave  = threadIdx.x >> 5;              // 0..7
    const int ngrp  = blockIdx.x >> 4;               // 0..7  -> n0 = ngrp*64
    const int m2    = (blockIdx.x & 15) * 8 + wave;  // 0..127 (32-row group)
    const int g     = lane >> 4;                     // K half selector
    const int l16   = lane & 15;
    const int nbase = ngrp * 64;

    v8f acc[2][4];
#pragma unroll
    for (int u = 0; u < 2; ++u)
#pragma unroll
        for (int t = 0; t < 4; ++t) acc[u][t] = (v8f){0,0,0,0,0,0,0,0};

    // Two A rows handled by this lane: m2*32 + l16 (+16)
    const float* __restrict__ arow0 = seq + (size_t)(m2 * 32 + l16) * E_DIM;
    const float* __restrict__ arow1 = arow0 + (size_t)16 * E_DIM;

    for (int kt = 0; kt < KTILES; ++kt) {
        // ---- two A fragments 16x32: slots 0..7 -> e = kt*32+g*8+s; 8..15 -> +16
        v16bf a_hi[2], a_lo[2];
#pragma unroll
        for (int u = 0; u < 2; ++u) {
            const float* p0 = (u ? arow1 : arow0) + kt * 32 + g * 8;
            const float* p1 = p0 + 16;
#pragma unroll
            for (int s = 0; s < 8; ++s) {
                float x = p0[s];
                __bf16 h = (__bf16)x;
                a_hi[u][s] = h;
                a_lo[u][s] = (__bf16)(x - (float)h);
                float y = p1[s];
                __bf16 h2 = (__bf16)y;
                a_hi[u][8 + s] = h2;
                a_lo[u][8 + s] = (__bf16)(y - (float)h2);
            }
        }
#pragma unroll
        for (int t = 0; t < 4; ++t) {
            const int nt = ngrp * 4 + t;
            const size_t fbase = (((size_t)kt * NTILES + nt) * 32 + lane) * 16;
            v16bf b_hi = *(const v16bf*)(whi + fbase);   // 32B coalesced
            v16bf b_lo = *(const v16bf*)(wlo + fbase);
#pragma unroll
            for (int u = 0; u < 2; ++u) {
                // (A_hi + A_lo)(B_hi + B_lo) ~= hi*hi + lo*hi + hi*lo
                acc[u][t] = __builtin_amdgcn_wmma_f32_16x16x32_bf16(
                    false, a_hi[u], false, b_hi, (short)0, acc[u][t], false, false);
                acc[u][t] = __builtin_amdgcn_wmma_f32_16x16x32_bf16(
                    false, a_lo[u], false, b_hi, (short)0, acc[u][t], false, false);
                acc[u][t] = __builtin_amdgcn_wmma_f32_16x16x32_bf16(
                    false, a_hi[u], false, b_lo, (short)0, acc[u][t], false, false);
            }
        }
    }

    // ---- store + bias: acc[u][t][v] -> row m2*32 + u*16 + v + 8*g,
    //                                    col nbase + t*16 + l16
#pragma unroll
    for (int t = 0; t < 4; ++t) {
        const int n  = nbase + t * 16 + l16;
        const float bv = bias[n];
#pragma unroll
        for (int u = 0; u < 2; ++u) {
#pragma unroll
            for (int v = 0; v < 8; ++v) {
                const int row = m2 * 32 + u * 16 + v + 8 * g;
                red[(size_t)row * RED + n] = acc[u][t][v] + bv;
            }
        }
    }
}

// ---------------------------------------------------------------------------
// Kernel 2: logits[r] = dot(seq[r,:], W[:,512]) + bias[512]   (column d=512)
// ---------------------------------------------------------------------------
__global__ __launch_bounds__(256)
void wpse_logits(const float* __restrict__ seq,
                 const float* __restrict__ W,
                 const float* __restrict__ bias,
                 float* __restrict__ logits)
{
    const int lane = threadIdx.x & 31;
    const int wave = threadIdx.x >> 5;
    const int row  = blockIdx.x * 8 + wave;       // 0..4095
    const float* p = seq + (size_t)row * E_DIM;
    float acc = 0.0f;
#pragma unroll 4
    for (int e = lane; e < E_DIM; e += 32)
        acc += p[e] * W[(size_t)e * WCOLS + RED];
#pragma unroll
    for (int off = 16; off > 0; off >>= 1)
        acc += __shfl_xor(acc, off, 32);
    if (lane == 0) logits[row] = acc + bias[RED];
}

// ---------------------------------------------------------------------------
// Kernel 3: per-span masked softmax + attention pooling + width-emb concat.
// ---------------------------------------------------------------------------
__global__ __launch_bounds__(256)
void wpse_span(const float* __restrict__ red,
               const float* __restrict__ logits,
               const int*   __restrict__ span_idx,
               const int*   __restrict__ span_mask,
               const float* __restrict__ width_emb,
               float*       __restrict__ out)
{
    __shared__ float s_attn[MAXW];
    __shared__ int   s_row[MAXW];

    const int span  = blockIdx.x;            // 0..4095
    const int tid   = threadIdx.x;
    const int b     = span / NSPANS;
    const int start = span_idx[span * 2 + 0];
    const int end   = span_idx[span * 2 + 1];
    const int width = end - start;

    if (tid < 32) {
        const int j = tid;
        const int raw = start + j;
        const bool valid = (j <= width) && (raw < T_DIM) && (raw >= 0);
        const int row = valid ? (b * T_DIM + raw) : (b * T_DIM);  // safe row
        s_row[j] = row;
        float l = valid ? logits[row] : NEG_INF_F;
        float mx = l;
#pragma unroll
        for (int off = 16; off > 0; off >>= 1)
            mx = fmaxf(mx, __shfl_xor(mx, off, 32));
        float e = valid ? __expf(l - mx) : 0.0f;
        float s = e;
#pragma unroll
        for (int off = 16; off > 0; off >>= 1)
            s += __shfl_xor(s, off, 32);
        s_attn[j] = e / s;                   // j=0 always valid -> s > 0
    }
    __syncthreads();

    const float maskf = span_mask[span] ? 1.0f : 0.0f;
    const size_t obase = (size_t)span * OUT_D;

    // attended[d] = sum_j attn[j] * red[row_j, d]   (coalesced over d)
    for (int d = tid; d < RED; d += 256) {
        float acc = 0.0f;
#pragma unroll
        for (int j = 0; j < MAXW; ++j)
            acc += s_attn[j] * red[(size_t)s_row[j] * RED + d];
        out[obase + d] = acc * maskf;
    }

    // width embedding concat
    if (tid < WEMB_D) {
        int w = width;
        w = (w < 0) ? 0 : ((w >= WEMB_N) ? (WEMB_N - 1) : w);
        out[obase + RED + tid] = width_emb[(size_t)w * WEMB_D + tid] * maskf;
    }
}

// ---------------------------------------------------------------------------
extern "C" void kernel_launch(void* const* d_in, const int* in_sizes, int n_in,
                              void* d_out, int out_size, void* d_ws, size_t ws_size,
                              hipStream_t stream)
{
    const float* seq      = (const float*)d_in[0];  // (B,T,E) f32
    const int*   span_idx = (const int*)  d_in[1];  // (B,NSPANS,2) int32
    const int*   mask     = (const int*)  d_in[2];  // (B,NSPANS) bool->int
    const float* W        = (const float*)d_in[3];  // (E, 513) f32
    const float* bias     = (const float*)d_in[4];  // (513,) f32
    const float* wemb     = (const float*)d_in[5];  // (64,128) f32
    float*       out      = (float*)d_out;          // (B,NSPANS,640) f32

    const int nrows = B_DIM * T_DIM;                // 4096

    // Workspace layout (all offsets 32B-aligned):
    //   red    : 4096*512 f32                     (8.39 MB)
    //   logits : 4096 f32                         (16 KB)
    //   whi/wlo: 524288 bf16 each                 (1 MB each)
    float*  red    = (float*)d_ws;
    float*  logits = red + (size_t)nrows * RED;
    __bf16* whi    = (__bf16*)(logits + nrows);
    __bf16* wlo    = whi + WFRAG_ELEMS;

    wpse_prep_w   <<<WFRAG_ELEMS / 256, 256, 0, stream>>>(W, whi, wlo);
    wpse_gemm_wmma<<<128, 256, 0, stream>>>(seq, whi, wlo, bias, red);
    wpse_logits   <<<nrows / 8, 256, 0, stream>>>(seq, W, bias, logits);
    wpse_span     <<<B_DIM * NSPANS, 256, 0, stream>>>(red, logits, span_idx,
                                                       mask, wemb, out);
}
